// LSTM_Dec_v1_31112743092807
// MI455X (gfx1250) — matrix operand order repaired and checked by reading, not compiled, not run on hardware
//
#include <hip/hip_runtime.h>
#include <hip/hip_bf16.h>

// ---------------------------------------------------------------------------
// LSTM decoder on MI455X (gfx1250), bf16 WMMA + async global->LDS staging.
//   SEQ_LEN=256 steps, BATCH=256, HIDDEN=1024, OUT=512.
//   Per step: gates = [x|h] @ Wcat^T + bias   (256x4096, K=1536)
//             cell update (sigmoid/tanh)       elementwise
//             o_t = c_new @ W_out^T + b_out    (256x512,  K=1024)
// Both GEMMs: block = 8 waves (8 M-tiles) sharing one 16-col N-slice; B tiles
// staged into LDS with global_load_async_to_lds_b128 (ASYNCcnt), double-
// buffered with a branch-free unrolled steady loop + peeled epilogue.
// ---------------------------------------------------------------------------

#define SEQ_LEN 256
#define HIDDEN  1024
#define OUTDIM  512
#define BATCH   256
#define KCAT    1536   /* OUTDIM + HIDDEN */

typedef __attribute__((ext_vector_type(16))) __bf16 v16bf;
typedef __attribute__((ext_vector_type(8)))  float  v8f;

union Frag16 { uint4 u[2]; v16bf v; };

// A-matrix 16x32 bf16 fragment (ISA 7.12.2): lane 0-15 = row M=lane,
// elems 0..7 = K+0..7, elems 8..15 = K+16..23; lanes 16-31: K offset +8.
__device__ __forceinline__ v16bf load_A_frag(const __bf16* __restrict__ A,
                                             int lda, int mbase, int k) {
    int lane = threadIdx.x & 31;
    const __bf16* p = A + (size_t)(mbase + (lane & 15)) * lda
                        + (k + ((lane >> 4) << 3));
    Frag16 f;
    f.u[0] = *reinterpret_cast<const uint4*>(p);
    f.u[1] = *reinterpret_cast<const uint4*>(p + 16);
    return f.v;
}

// B-matrix 32x16 bf16 fragment sourced from an LDS tile laid out as
// lds[g][n][0..31] (32 bf16 of K per row) -> ds_load_b128 x2 per lane.
__device__ __forceinline__ v16bf lds_B_frag(const __bf16* ldsW, int g) {
    int lane = threadIdx.x & 31;
    const __bf16* p = ldsW + g * 512 + ((lane & 15) << 5) + ((lane >> 4) << 3);
    Frag16 f;
    f.u[0] = *reinterpret_cast<const uint4*>(p);
    f.u[1] = *reinterpret_cast<const uint4*>(p + 16);
    return f.v;
}

// Out-proj B fragment from an LDS slab lds[n][0..127] (128 K per row).
__device__ __forceinline__ v16bf lds_W_frag(const __bf16* ldsW, int kk) {
    int lane = threadIdx.x & 31;
    const __bf16* p = ldsW + ((lane & 15) << 7) + kk + ((lane >> 4) << 3);
    Frag16 f;
    f.u[0] = *reinterpret_cast<const uint4*>(p);
    f.u[1] = *reinterpret_cast<const uint4*>(p + 16);
    return f.v;
}

// CDNA5 async global->LDS copy, 16B per lane, tracked by ASYNCcnt.
// LDS aperture keeps the byte offset in the low 32 bits of the flat address.
__device__ __forceinline__ void async_copy_b128(void* ldsDst, const void* gSrc) {
    unsigned int ldsOff = (unsigned int)(unsigned long long)ldsDst;
    asm volatile("global_load_async_to_lds_b128 %0, %1, off"
                 :: "v"(ldsOff), "v"(gSrc) : "memory");
}
__device__ __forceinline__ void wait_async_le1() {
    asm volatile("s_wait_asynccnt 0x1" ::: "memory");
}
__device__ __forceinline__ void wait_async_0() {
    asm volatile("s_wait_asynccnt 0x0" ::: "memory");
}

__device__ __forceinline__ float sigmoid_fast(float x) {
    return 1.0f / (1.0f + __expf(-x));
}
__device__ __forceinline__ float tanh_fast(float x) {
    float e = __expf(2.0f * x);
    return (e - 1.0f) / (e + 1.0f);
}

// ---------------------------------------------------------------------------
// Prep: Wcat(4096x1536) = bf16[W_ih | W_hh]; WoutBf(512x1024); bias = b_ih+b_hh
// ---------------------------------------------------------------------------
__global__ __launch_bounds__(256)
void prep_weights(const float* __restrict__ Wih, const float* __restrict__ Whh,
                  const float* __restrict__ Wout, const float* __restrict__ bih,
                  const float* __restrict__ bhh,
                  __bf16* __restrict__ Wcat, __bf16* __restrict__ WoutBf,
                  float* __restrict__ bias) {
    size_t i = (size_t)blockIdx.x * blockDim.x + threadIdx.x;
    const size_t nWcat = (size_t)4 * HIDDEN * KCAT;          // 6,291,456
    const size_t nWout = (size_t)OUTDIM * HIDDEN;            //   524,288
    if (i < nWcat) {
        size_t j = i / KCAT, k = i % KCAT;
        float v = (k < OUTDIM) ? Wih[j * OUTDIM + k]
                               : Whh[j * HIDDEN + (k - OUTDIM)];
        Wcat[i] = (__bf16)v;
    } else if (i < nWcat + nWout) {
        size_t r = i - nWcat;
        WoutBf[r] = (__bf16)Wout[r];
    } else if (i < nWcat + nWout + (size_t)4 * HIDDEN) {
        size_t r = i - nWcat - nWout;
        bias[r] = bih[r] + bhh[r];
    }
}

// ---------------------------------------------------------------------------
// Init state: A0 = [x0=0 | bf16(c_vector)], cstate = c_vector, cbf = bf16(c)
// ---------------------------------------------------------------------------
__global__ __launch_bounds__(256)
void init_state(const float* __restrict__ cvec, __bf16* __restrict__ A0,
                float* __restrict__ cstate, __bf16* __restrict__ cbf) {
    size_t i = (size_t)blockIdx.x * blockDim.x + threadIdx.x;
    const size_t nA = (size_t)BATCH * KCAT;     // 393,216
    const size_t nC = (size_t)BATCH * HIDDEN;   // 262,144
    if (i < nA) {
        size_t b = i / KCAT, k = i % KCAT;
        A0[i] = (k < OUTDIM) ? (__bf16)0.0f
                             : (__bf16)cvec[b * HIDDEN + (k - OUTDIM)];
    } else if (i < nA + nC) {
        size_t r = i - nA;
        float v = cvec[r];
        cstate[r] = v;
        cbf[r]    = (__bf16)v;
    }
}

// ---------------------------------------------------------------------------
// Step kernel 1: gates GEMM + cell update.
// Grid: 128 blocks = 2 M-blocks x 64 N-tiles; block = 8 waves = 8 M-tiles.
// Per 32-K stage, the 256 threads async-copy 4KB of B (i,f,g,o tiles).
// Steady loop unrolled over both LDS buffers -> constant addresses, no
// branches; epilogue peeled for the last two stages.
// ---------------------------------------------------------------------------
__device__ __forceinline__ void gates_consume(const __bf16* ldsbuf, v16bf a,
                                              v8f& ai, v8f& af,
                                              v8f& ag, v8f& ao) {
    v16bf b0 = lds_B_frag(ldsbuf, 0);
    ai = __builtin_amdgcn_wmma_f32_16x16x32_bf16(
             false, a, false, b0, (short)0, ai, false, false);
    v16bf b1 = lds_B_frag(ldsbuf, 1);
    af = __builtin_amdgcn_wmma_f32_16x16x32_bf16(
             false, a, false, b1, (short)0, af, false, false);
    v16bf b2 = lds_B_frag(ldsbuf, 2);
    ag = __builtin_amdgcn_wmma_f32_16x16x32_bf16(
             false, a, false, b2, (short)0, ag, false, false);
    v16bf b3 = lds_B_frag(ldsbuf, 3);
    ao = __builtin_amdgcn_wmma_f32_16x16x32_bf16(
             false, a, false, b3, (short)0, ao, false, false);
}

__global__ __launch_bounds__(256)
void lstm_gates_cell(const __bf16* __restrict__ Wcat,
                     const float*  __restrict__ bias,
                     const __bf16* __restrict__ Ain,   // [x|h] step t (B x 1536)
                     __bf16*       __restrict__ Aout,  // h written for step t+1
                     float*        __restrict__ cstate,
                     __bf16*       __restrict__ cbf,
                     float*        __restrict__ hs,
                     float*        __restrict__ cs,
                     int t) {
    __shared__ __bf16 ldsB[2][4 * 16 * 32];   // 2 stages x (4 gates x 16n x 32k)

    const int tid   = threadIdx.x;
    const int lane  = tid & 31;
    const int wv    = tid >> 5;                       // wave 0..7
    const int ntile = blockIdx.x & 63;
    const int mblk  = blockIdx.x >> 6;                // 0..1
    const int mbase = (mblk << 7) + (wv << 4);        // 8 M-tiles per block
    const int nbase = ntile << 4;

    // Per-thread slice of one 4KB B stage: g = tid>>6, row r, 8-elem seg.
    const int sg = tid >> 6;
    const int sr = (tid >> 2) & 15;
    const int ss = tid & 3;
    const __bf16* stageSrc = Wcat + (size_t)(sg * HIDDEN + nbase + sr) * KCAT
                                  + (ss << 3);
    const int stageDstOff = (sg << 9) + (sr << 5) + (ss << 3);

    v8f acc_i = {}, acc_f = {}, acc_g = {}, acc_o = {};

    // Prologue: two stages in flight, first A fragment in registers.
    async_copy_b128(&ldsB[0][stageDstOff], stageSrc);
    async_copy_b128(&ldsB[1][stageDstOff], stageSrc + 32);
    v16bf a = load_A_frag(Ain, KCAT, mbase, 0);

#pragma unroll 1
    for (int k = 0; k < KCAT - 64; k += 64) {
        // ---- stage k (buffer 0) ----
        wait_async_le1();                 // stage k landed (oldest of 2)
        __syncthreads();
        v16bf a1 = load_A_frag(Ain, KCAT, mbase, k + 32);
        gates_consume(ldsB[0], a, acc_i, acc_f, acc_g, acc_o);
        __syncthreads();                  // all waves done reading buffer 0
        async_copy_b128(&ldsB[0][stageDstOff], stageSrc + (k + 64));
        // ---- stage k+32 (buffer 1) ----
        wait_async_le1();                 // stage k+32 landed
        __syncthreads();
        v16bf a2 = load_A_frag(Ain, KCAT, mbase, k + 64);
        gates_consume(ldsB[1], a1, acc_i, acc_f, acc_g, acc_o);
        __syncthreads();                  // all waves done reading buffer 1
        async_copy_b128(&ldsB[1][stageDstOff], stageSrc + (k + 96));
        a = a2;
    }
    // Epilogue: stages KCAT-64 (buf0) and KCAT-32 (buf1) still in flight.
    wait_async_le1();
    __syncthreads();
    v16bf a1 = load_A_frag(Ain, KCAT, mbase, KCAT - 32);
    gates_consume(ldsB[0], a, acc_i, acc_f, acc_g, acc_o);
    wait_async_0();
    __syncthreads();
    gates_consume(ldsB[1], a1, acc_i, acc_f, acc_g, acc_o);

    // C/D layout: elem v <-> row mbase + v + 8*(lane>>4), col nbase + (lane&15)
    const int colN = lane & 15;
    const int rsel = (lane >> 4) << 3;
    const int j    = nbase + colN;
    const float b_i = bias[0 * HIDDEN + j];
    const float b_f = bias[1 * HIDDEN + j];
    const float b_g = bias[2 * HIDDEN + j];
    const float b_o = bias[3 * HIDDEN + j];

#pragma unroll
    for (int v = 0; v < 8; ++v) {
        int m = mbase + v + rsel;
        float gi = sigmoid_fast(acc_i[v] + b_i);
        float gf = sigmoid_fast(acc_f[v] + b_f);
        float gg = tanh_fast(acc_g[v] + b_g);
        float go = sigmoid_fast(acc_o[v] + b_o);
        size_t ci  = (size_t)m * HIDDEN + j;
        float cold = cstate[ci];
        float cnew = gf * cold + gi * gg;
        float hnew = go * tanh_fast(cnew);
        cstate[ci] = cnew;
        cbf[ci]    = (__bf16)cnew;
        Aout[(size_t)m * KCAT + OUTDIM + j] = (__bf16)hnew;
        size_t oi = (size_t)m * ((size_t)SEQ_LEN * HIDDEN)
                  + (size_t)t * HIDDEN + j;
        hs[oi] = hnew;
        cs[oi] = cnew;
    }
}

// ---------------------------------------------------------------------------
// Step kernel 2: o_t = c_new @ W_out^T + b_out; also writes bf16 x for t+1.
// Grid: 64 blocks = 2 M-blocks x 32 N-tiles; block = 8 waves = 8 M-tiles.
// B staged in 128-K slabs (4KB = 256 threads x 16B), double-buffered async.
// ---------------------------------------------------------------------------
__device__ __forceinline__ void outproj_consume(const __bf16* ldsbuf,
                                                const __bf16* __restrict__ cbf,
                                                int mbase, int kcbase, v8f& acc) {
#pragma unroll
    for (int kk = 0; kk < 128; kk += 32) {
        v16bf a = load_A_frag(cbf, HIDDEN, mbase, kcbase + kk);
        v16bf b = lds_W_frag(ldsbuf, kk);
        acc = __builtin_amdgcn_wmma_f32_16x16x32_bf16(
                  false, a, false, b, (short)0, acc, false, false);
    }
}

__global__ __launch_bounds__(256)
void lstm_outproj(const __bf16* __restrict__ WoutBf,
                  const float*  __restrict__ b_out,
                  const __bf16* __restrict__ cbf,
                  __bf16*       __restrict__ Aout,   // x part for step t+1
                  float*        __restrict__ outs,
                  int t) {
    __shared__ __bf16 ldsW[2][16 * 128];   // 2 stages x (16 n x 128 k) = 8KB

    const int tid   = threadIdx.x;
    const int lane  = tid & 31;
    const int wv    = tid >> 5;
    const int ntile = blockIdx.x & 31;
    const int mblk  = blockIdx.x >> 5;                // 0..1
    const int mbase = (mblk << 7) + (wv << 4);
    const int nbase = ntile << 4;

    // Per-thread slice of one 4KB stage: row r = tid>>4, 8-elem seg = tid&15.
    const int sr = tid >> 4;
    const int ss = tid & 15;
    const __bf16* stageSrc = WoutBf + (size_t)(nbase + sr) * HIDDEN + (ss << 3);
    const int stageDstOff = (sr << 7) + (ss << 3);

    v8f acc = {};

    async_copy_b128(&ldsW[0][stageDstOff], stageSrc);
    async_copy_b128(&ldsW[1][stageDstOff], stageSrc + 128);

#pragma unroll 1
    for (int kc = 0; kc < HIDDEN - 256; kc += 256) {
        // gfx1250 global_prefetch_b8 on the A rows of the next chunk
        __builtin_prefetch(cbf + (size_t)(mbase + (lane & 15)) * HIDDEN
                               + (kc + 128), 0, 1);
        // ---- stage kc (buffer 0) ----
        wait_async_le1();
        __syncthreads();
        outproj_consume(ldsW[0], cbf, mbase, kc, acc);
        __syncthreads();
        async_copy_b128(&ldsW[0][stageDstOff], stageSrc + (kc + 256));
        // ---- stage kc+128 (buffer 1) ----
        wait_async_le1();
        __syncthreads();
        outproj_consume(ldsW[1], cbf, mbase, kc + 128, acc);
        __syncthreads();
        async_copy_b128(&ldsW[1][stageDstOff], stageSrc + (kc + 384));
    }
    // Epilogue: stages HIDDEN-256 (buf0) and HIDDEN-128 (buf1) in flight.
    wait_async_le1();
    __syncthreads();
    outproj_consume(ldsW[0], cbf, mbase, HIDDEN - 256, acc);
    wait_async_0();
    __syncthreads();
    outproj_consume(ldsW[1], cbf, mbase, HIDDEN - 128, acc);

    int colN = lane & 15;
    int rsel = (lane >> 4) << 3;
    int j    = nbase + colN;
    float bo = b_out[j];

#pragma unroll
    for (int v = 0; v < 8; ++v) {
        int m = mbase + v + rsel;
        float val = acc[v] + bo;
        outs[(size_t)m * ((size_t)SEQ_LEN * OUTDIM) + (size_t)t * OUTDIM + j] = val;
        Aout[(size_t)m * KCAT + j] = (__bf16)val;
    }
}

// ---------------------------------------------------------------------------
// Host launcher. Inputs (setup_inputs order):
//   0 c_vector (B,H) f32      1 W_ih (4H,OUT) f32   2 W_hh (4H,H) f32
//   3 b_ih (4H) f32           4 b_hh (4H) f32
//   5 W_out (OUT,H) f32       6 b_out (OUT) f32
// Output: hs (B,T,H) | cs (B,T,H) | outs (B,T,OUT), f32 concat.
// ---------------------------------------------------------------------------
extern "C" void kernel_launch(void* const* d_in, const int* in_sizes, int n_in,
                              void* d_out, int out_size, void* d_ws, size_t ws_size,
                              hipStream_t stream) {
    const float* cvec = (const float*)d_in[0];
    const float* Wih  = (const float*)d_in[1];
    const float* Whh  = (const float*)d_in[2];
    const float* bih  = (const float*)d_in[3];
    const float* bhh  = (const float*)d_in[4];
    const float* Wout = (const float*)d_in[5];
    const float* bout = (const float*)d_in[6];

    float* hs   = (float*)d_out;
    float* cs   = hs + (size_t)BATCH * SEQ_LEN * HIDDEN;
    float* outs = cs + (size_t)BATCH * SEQ_LEN * HIDDEN;

    // Workspace carve-up (~16.8 MB total)
    char* ws = (char*)d_ws;
    __bf16* Wcat   = (__bf16*)ws;  ws += (size_t)4 * HIDDEN * KCAT * sizeof(__bf16);
    __bf16* WoutBf = (__bf16*)ws;  ws += (size_t)OUTDIM * HIDDEN * sizeof(__bf16);
    float*  bias   = (float*)ws;   ws += (size_t)4 * HIDDEN * sizeof(float);
    __bf16* Abuf0  = (__bf16*)ws;  ws += (size_t)BATCH * KCAT * sizeof(__bf16);
    __bf16* Abuf1  = (__bf16*)ws;  ws += (size_t)BATCH * KCAT * sizeof(__bf16);
    float*  cstate = (float*)ws;   ws += (size_t)BATCH * HIDDEN * sizeof(float);
    __bf16* cbf    = (__bf16*)ws;
    __bf16* Abuf[2] = {Abuf0, Abuf1};

    // 6,291,456 + 524,288 + 4,096 = 6,819,840 elems -> 26,640 blocks of 256
    prep_weights<<<26640, 256, 0, stream>>>(Wih, Whh, Wout, bih, bhh,
                                            Wcat, WoutBf, bias);
    // 393,216 + 262,144 = 655,360 elems -> 2,560 blocks of 256
    init_state<<<2560, 256, 0, stream>>>(cvec, Abuf[0], cstate, cbf);

    for (int t = 0; t < SEQ_LEN; ++t) {
        const __bf16* Ain  = Abuf[t & 1];
        __bf16*       Anew = Abuf[(t + 1) & 1];
        lstm_gates_cell<<<128, 256, 0, stream>>>(Wcat, bias, Ain, Anew,
                                                 cstate, cbf, hs, cs, t);
        lstm_outproj<<<64, 256, 0, stream>>>(WoutBf, bout, cbf, Anew, outs, t);
    }
}
